// RadNet_31190052504103
// MI455X (gfx1250) — matrix operand order repaired and checked by reading, not compiled, not run on hardware
//
#include <hip/hip_runtime.h>
#include <hip/hip_bf16.h>

// CDNA5 (gfx1250) wave32 WMMA types
typedef __attribute__((ext_vector_type(16))) _Float16 v16h;
typedef __attribute__((ext_vector_type(8)))  _Float16 v8h;
typedef __attribute__((ext_vector_type(8)))  float    v8f;

#define N_IMG  512
#define N_NBR  48
#define N_MOL  16
#define CUTV   0.926f          // 1.852/2
#define INV2S2 12.5f           // 0.5 / 0.2^2
#define CONV_WAVES 4

union AFrag { v16h v; v8h h[2]; };

// ---------------------------------------------------------------------------
// Zero-fill (halo + channel padding must be zero; cleared once per launch)
// ---------------------------------------------------------------------------
__global__ void clear_kernel(float4* __restrict__ p, long n16) {
  long i = (long)blockIdx.x * blockDim.x + threadIdx.x;
  const long stride = (long)gridDim.x * blockDim.x;
  const float4 z = make_float4(0.f, 0.f, 0.f, 0.f);
  for (; i < n16; i += stride) p[i] = z;
}

// ---------------------------------------------------------------------------
// Conv weight pack: f32 [Cout][Cin][27] -> f16 [27][CoutP][CinP]
// ---------------------------------------------------------------------------
__global__ void pack_convw_kernel(const float* __restrict__ src,
                                  _Float16* __restrict__ dst,
                                  int Cout, int Cin, int CoutP, int CinP) {
  const int total = 27 * CoutP * CinP;
  int i = blockIdx.x * 256 + threadIdx.x;
  if (i >= total) return;
  const int tap = i / (CoutP * CinP);
  const int r   = i - tap * CoutP * CinP;
  const int co  = r / CinP;
  const int c   = r - co * CinP;
  const float v = (co < Cout && c < Cin) ? src[(co * Cin + c) * 27 + tap] : 0.0f;
  dst[i] = (_Float16)v;
}

// ---------------------------------------------------------------------------
// FC weight pack: f32 [M][K] -> f16 [MP][K] (zero pad rows)
// ---------------------------------------------------------------------------
__global__ void pack_fcw_kernel(const float* __restrict__ src,
                                _Float16* __restrict__ dst,
                                int M, int K, int MP) {
  const size_t total = (size_t)MP * K;
  size_t i = (size_t)blockIdx.x * 256 + threadIdx.x;
  if (i >= total) return;
  const int m = (int)(i / K);
  const int k = (int)(i - (size_t)m * K);
  dst[i] = (_Float16)((m < M) ? src[(size_t)m * K + k] : 0.0f);
}

// ---------------------------------------------------------------------------
// FC1 weight pack with flatten-order permutation:
// reference flatten k = c*512 + p  ->  our channels-last k' = p*32 + c
// ---------------------------------------------------------------------------
__global__ void pack_fc1w_kernel(const float* __restrict__ src,
                                 _Float16* __restrict__ dst) {
  size_t i = (size_t)blockIdx.x * 256 + threadIdx.x;
  if (i >= (size_t)1024 * 16384) return;
  const int m = (int)(i >> 14);
  const int r = (int)(i & 16383);
  const int p = r >> 5;
  const int c = r & 31;
  dst[i] = (_Float16)src[(size_t)m * 16384 + c * 512 + p];
}

// ---------------------------------------------------------------------------
// Kernel 1: Gaussian splat -> channels-last padded grid [N][17^3][32], ch 0
// ---------------------------------------------------------------------------
__global__ void splat_kernel(const float* __restrict__ pos,
                             const float* __restrict__ Z,
                             const float* __restrict__ nbr,
                             const float* __restrict__ use_nbr,
                             _Float16* __restrict__ out) {
  __shared__ float sdx[N_NBR], sdy[N_NBR], sdz[N_NBR], sw[N_NBR];
  const int n = blockIdx.x;
  const int t = threadIdx.x;
  if (t < N_NBR) {
    const float px = pos[n * 3 + 0], py = pos[n * 3 + 1], pz = pos[n * 3 + 2];
    sdx[t] = px - nbr[(n * N_NBR + t) * 3 + 0];
    sdy[t] = py - nbr[(n * N_NBR + t) * 3 + 1];
    sdz[t] = pz - nbr[(n * N_NBR + t) * 3 + 2];
    sw[t]  = Z[n] * use_nbr[n * N_NBR + t];
  }
  __syncthreads();
  const float step = (2.0f * CUTV) / 14.0f;
  for (int g = t; g < 3375; g += blockDim.x) {
    const int ix = g / 225;
    const int iy = (g / 15) % 15;
    const int iz = g % 15;
    const float gx = -CUTV + ix * step;
    const float gy = -CUTV + iy * step;
    const float gz = -CUTV + iz * step;
    float acc = 0.0f;
    #pragma unroll 4
    for (int k = 0; k < N_NBR; ++k) {
      const float dx = sdx[k] - gx;
      const float dy = sdy[k] - gy;
      const float dz = sdz[k] - gz;
      acc += sw[k] * __expf(-(dx * dx + dy * dy + dz * dz) * INV2S2);
    }
    const float r = sqrtf(gx * gx + gy * gy + gz * gz);
    const float filt = 0.5f * erfcf(r - CUTV);
    const size_t spatial = ((size_t)(ix + 1) * 17 + (iy + 1)) * 17 + (iz + 1);
    out[((size_t)n * 4913 + spatial) * 32] = (_Float16)(acc * filt);
  }
}

// ---------------------------------------------------------------------------
// Kernel 2: Conv3D 3x3x3 (pad 1) + bias + ELU, channels-last implicit GEMM.
// in : f16 [img][DP^3][CINP]   (halo-padded, channel-padded, zeros outside)
// w  : f16 [27][COUTP][CINP]
// out: f16 [img][DPO^3][COUTP] (padOut) or [img][DOUT^3][COUTP] (compact)
// 27 shifted GEMMs over K = channels; every load/store is b128.
// ---------------------------------------------------------------------------
template<int CINP, int COUTP, int STRIDE, int DIN, int DOUT>
__global__ __launch_bounds__(32 * CONV_WAVES)
void conv3d_wmma_kernel(const _Float16* __restrict__ in,
                        const _Float16* __restrict__ wpack,
                        const float* __restrict__ bias,
                        _Float16* __restrict__ out,
                        int Cout, int padOut) {
  constexpr int DP   = DIN + 2;
  constexpr int DP3  = DP * DP * DP;
  constexpr int DPO  = DOUT + 2;
  constexpr int POUT = DOUT * DOUT * DOUT;

  const int lane = threadIdx.x & 31;
  const int wave = threadIdx.x >> 5;
  const int img  = blockIdx.z;
  const int co0  = blockIdx.y * 16;
  const int tile = blockIdx.x * CONV_WAVES + wave;

  const int hi  = lane >> 4;
  const int l15 = lane & 15;

  const int  pcol   = tile * 16 + l15;
  const bool pvalid = (pcol < POUT);
  const int  pc     = pvalid ? pcol : (POUT - 1);
  const int  oz  = pc / (DOUT * DOUT);
  const int  rem = pc - oz * DOUT * DOUT;
  const int  oy  = rem / DOUT;
  const int  ox  = rem - oy * DOUT;

  // per-lane base pointers; all tap shifts are compile-time immediates
  const _Float16* bcol = in +
      ((size_t)img * DP3 + ((size_t)(oz * STRIDE) * DP + oy * STRIDE) * DP + ox * STRIDE) * CINP
      + (hi << 4);
  const _Float16* arow = wpack + (size_t)(co0 + l15) * CINP + (hi << 4);

  v8f acc = {};
  #pragma unroll
  for (int kd = 0; kd < 3; ++kd)
    #pragma unroll
    for (int kh = 0; kh < 3; ++kh)
      #pragma unroll
      for (int kw = 0; kw < 3; ++kw) {
        const int tap = (kd * 3 + kh) * 3 + kw;
        const _Float16* bp = bcol + ((kd * DP + kh) * DP + kw) * CINP;
        const _Float16* ap = arow + (size_t)tap * (COUTP * CINP);
        #pragma unroll
        for (int c0 = 0; c0 < CINP; c0 += 32) {
          AFrag a, b;
          a.h[0] = *(const v8h*)(ap + c0);          // A halves 0..7
          a.h[1] = *(const v8h*)(ap + c0 + 8);      // A halves 8..15
          b.h[0] = *(const v8h*)(bp + c0);          // B halves 0..7
          b.h[1] = *(const v8h*)(bp + c0 + 8);      // B halves 8..15
          acc = __builtin_amdgcn_wmma_f32_16x16x32_f16(false, a.v, false, b.v,
                                                       (short)0, acc, false, false);
        }
      }

  // Epilogue: 8 consecutive output channels per lane -> one b128 store.
  if (pvalid) {
    v8h ov;
    #pragma unroll
    for (int r = 0; r < 8; ++r) {
      const int co = co0 + (hi << 3) + r;           // M = r + 8*hi
      const float bv = bias[(co < Cout) ? co : 0];
      float v = acc[r] + bv;
      v = (v > 0.0f) ? v : (__expf(v) - 1.0f);      // ELU
      ov[r] = (_Float16)((co < Cout) ? v : 0.0f);   // zero-fill pad channels
    }
    size_t spatial;
    if (padOut)
      spatial = ((size_t)(oz + 1) * DPO + (oy + 1)) * DPO + (ox + 1);
    else
      spatial = (size_t)pc;
    const size_t base = padOut ? ((size_t)img * (DPO * DPO * DPO) + spatial)
                               : ((size_t)img * POUT + spatial);
    *(v8h*)(out + base * COUTP + co0 + (hi << 3)) = ov;
  }
}

// ---------------------------------------------------------------------------
// Kernel 3: FC on WMMA. x: f16 [N][K] (K%32==0), wpack: f16 [MP][K].
// ---------------------------------------------------------------------------
__global__ __launch_bounds__(32)
void fc_wmma_kernel(const _Float16* __restrict__ x,
                    const _Float16* __restrict__ wpack,
                    const float* __restrict__ bias,
                    _Float16* __restrict__ outH,
                    float* __restrict__ outF,
                    int M, int K, int applyElu, int useHalf) {
  const int lane = threadIdx.x;
  const int n0 = blockIdx.x * 16;
  const int m0 = blockIdx.y * 16;
  const int hi  = lane >> 4;
  const int l15 = lane & 15;

  const int ncol = n0 + l15;
  const _Float16* wrow = wpack + (size_t)(m0 + l15) * K + (hi << 4);
  const _Float16* xcol = x + (size_t)ncol * K + (hi << 4);

  v8f acc = {};
  for (int k0 = 0; k0 < K; k0 += 32) {
    AFrag a, b;
    a.h[0] = *(const v8h*)(wrow + k0);
    a.h[1] = *(const v8h*)(wrow + k0 + 8);
    b.h[0] = *(const v8h*)(xcol + k0);
    b.h[1] = *(const v8h*)(xcol + k0 + 8);
    acc = __builtin_amdgcn_wmma_f32_16x16x32_f16(false, a.v, false, b.v,
                                                 (short)0, acc, false, false);
  }

  #pragma unroll
  for (int r = 0; r < 8; ++r) {
    const int m = m0 + r + (hi ? 8 : 0);
    if (m < M) {
      float v = acc[r] + bias[m];
      if (applyElu) v = (v > 0.0f) ? v : (__expf(v) - 1.0f);
      if (useHalf) outH[(size_t)ncol * M + m] = (_Float16)v;
      else         outF[(size_t)ncol * M + m] = v;
    }
  }
}

// ---------------------------------------------------------------------------
// Kernel 4: segment_sum -> d_out [16][6] fp32
// ---------------------------------------------------------------------------
__global__ void segment_kernel(const float* __restrict__ fc2,
                               const int* __restrict__ index,
                               float* __restrict__ out) {
  const int t = threadIdx.x;
  if (t >= N_MOL * 6) return;
  const int m = t / 6;
  const int j = t - m * 6;
  float s = 0.0f;
  for (int a = 0; a < N_IMG; ++a)
    if (index[a] == m) s += fc2[a * 6 + j];
  out[t] = s;
}

// ---------------------------------------------------------------------------
// Host orchestration
// ---------------------------------------------------------------------------
static inline dim3 conv_grid(int Pout, int CoutP) {
  const int tiles = (Pout + 15) / 16;
  return dim3((tiles + CONV_WAVES - 1) / CONV_WAVES, CoutP / 16, N_IMG);
}

extern "C" void kernel_launch(void* const* d_in, const int* in_sizes, int n_in,
                              void* d_out, int out_size, void* d_ws, size_t ws_size,
                              hipStream_t stream) {
  (void)in_sizes; (void)n_in; (void)out_size; (void)ws_size;

  const float* pos     = (const float*)d_in[0];
  const float* Z       = (const float*)d_in[1];
  const float* nbr     = (const float*)d_in[2];
  const float* use_nbr = (const float*)d_in[3];
  const int*   index   = (const int*)d_in[4];

  char* ws = (char*)d_ws;
  // Distinct regions per (geometry x channel-stride) layout; halos never alias.
  const size_t S0 = (size_t)N_IMG * 4913 * 32 * 2;  // 15-grid padded, 32ch
  const size_t S1 = (size_t)N_IMG * 4913 * 64 * 2;  // 15-grid padded, 64ch
  const size_t S2 = (size_t)N_IMG * 1000 * 64 * 2;  // 8-grid padded, 64ch
  const size_t S3 = (size_t)N_IMG * 1000 * 32 * 2;  // 8-grid padded, 32ch
  const size_t S4 = (size_t)N_IMG * 512 * 32 * 2;   // 8-grid compact, 32ch
  _Float16* R0a = (_Float16*)(ws);
  _Float16* R0b = (_Float16*)(ws + S0);
  _Float16* R1a = (_Float16*)(ws + 2 * S0);
  _Float16* R1b = (_Float16*)(ws + 2 * S0 + S1);
  _Float16* R2  = (_Float16*)(ws + 2 * S0 + 2 * S1);
  _Float16* R3a = (_Float16*)(ws + 2 * S0 + 2 * S1 + S2);
  _Float16* R3b = (_Float16*)(ws + 2 * S0 + 2 * S1 + S2 + S3);
  _Float16* R4  = (_Float16*)(ws + 2 * S0 + 2 * S1 + S2 + 2 * S3);
  size_t off = 2 * S0 + 2 * S1 + S2 + 2 * S3 + S4;
  _Float16* fc1out = (_Float16*)(ws + off); off += (size_t)N_IMG * 1024 * 2;
  float*    fc2buf = (float*)(ws + off);    off += (size_t)N_IMG * 6 * 4;
  off = (off + 15) & ~(size_t)15;

  const size_t CLEAR_BYTES = 2 * S0 + 2 * S1 + S2 + 2 * S3; // padded regions only
  clear_kernel<<<8192, 256, 0, stream>>>((float4*)ws, (long)(CLEAR_BYTES / 16));

  // Pack conv weights: [27][CoutP][CinP] f16
  const int  cinA [11] = { 1,64,64,16,16,16,16,64,32,32,32};
  const int  coutA[11] = {64,64,16,16,16,16,64,32,32,32,32};
  const int  cinP [11] = {32,64,64,32,32,32,32,64,32,32,32};
  const int  coutP[11] = {64,64,32,32,32,32,64,32,32,32,32};
  _Float16* wp[11];
  for (int i = 0; i < 11; ++i) {
    const int total = 27 * coutP[i] * cinP[i];
    wp[i] = (_Float16*)(ws + off);
    off += (size_t)total * 2;
    pack_convw_kernel<<<(total + 255) / 256, 256, 0, stream>>>(
        (const float*)d_in[5 + 2 * i], wp[i], coutA[i], cinA[i], coutP[i], cinP[i]);
  }
  // FC weights (fc1 permuted for channels-last flatten)
  _Float16* fc1p = (_Float16*)(ws + off); off += (size_t)1024 * 16384 * 2;
  pack_fc1w_kernel<<<(int)(((size_t)1024 * 16384 + 255) / 256), 256, 0, stream>>>(
      (const float*)d_in[27], fc1p);
  _Float16* fc2p = (_Float16*)(ws + off); off += (size_t)16 * 1024 * 2;
  pack_fcw_kernel<<<(16 * 1024 + 255) / 256, 256, 0, stream>>>(
      (const float*)d_in[29], fc2p, 6, 1024, 16);

  // 1) splat -> R0a
  splat_kernel<<<N_IMG, 256, 0, stream>>>(pos, Z, nbr, use_nbr, R0a);

  // 2) conv stack (channels-last WMMA implicit GEMM)
  const float* B0  = (const float*)d_in[6];
  const float* B1  = (const float*)d_in[8];
  const float* B2  = (const float*)d_in[10];
  const float* B3  = (const float*)d_in[12];
  const float* B4  = (const float*)d_in[14];
  const float* B5  = (const float*)d_in[16];
  const float* B6  = (const float*)d_in[18];
  const float* B7  = (const float*)d_in[20];
  const float* B8  = (const float*)d_in[22];
  const float* B9  = (const float*)d_in[24];
  const float* B10 = (const float*)d_in[26];

  conv3d_wmma_kernel<32,64,1,15,15><<<conv_grid(3375,64),128,0,stream>>>(R0a, wp[0],  B0,  R1a, 64, 1);
  conv3d_wmma_kernel<64,64,1,15,15><<<conv_grid(3375,64),128,0,stream>>>(R1a, wp[1],  B1,  R1b, 64, 1);
  conv3d_wmma_kernel<64,32,1,15,15><<<conv_grid(3375,32),128,0,stream>>>(R1b, wp[2],  B2,  R0b, 16, 1);
  conv3d_wmma_kernel<32,32,1,15,15><<<conv_grid(3375,32),128,0,stream>>>(R0b, wp[3],  B3,  R0a, 16, 1);
  conv3d_wmma_kernel<32,32,1,15,15><<<conv_grid(3375,32),128,0,stream>>>(R0a, wp[4],  B4,  R0b, 16, 1);
  conv3d_wmma_kernel<32,32,1,15,15><<<conv_grid(3375,32),128,0,stream>>>(R0b, wp[5],  B5,  R0a, 16, 1);
  conv3d_wmma_kernel<32,64,2,15, 8><<<conv_grid( 512,64),128,0,stream>>>(R0a, wp[6],  B6,  R2,  64, 1);
  conv3d_wmma_kernel<64,32,1, 8, 8><<<conv_grid( 512,32),128,0,stream>>>(R2,  wp[7],  B7,  R3a, 32, 1);
  conv3d_wmma_kernel<32,32,1, 8, 8><<<conv_grid( 512,32),128,0,stream>>>(R3a, wp[8],  B8,  R3b, 32, 1);
  conv3d_wmma_kernel<32,32,1, 8, 8><<<conv_grid( 512,32),128,0,stream>>>(R3b, wp[9],  B9,  R3a, 32, 1);
  conv3d_wmma_kernel<32,32,1, 8, 8><<<conv_grid( 512,32),128,0,stream>>>(R3a, wp[10], B10, R4,  32, 0);
  // R4: compact channels-last [512][512][32] == [512][16384] with k = p*32+c

  // 3) FC1 (weights permuted to match k' = p*32+c), ELU, f16 out
  {
    dim3 grid(N_IMG / 16, 1024 / 16);
    fc_wmma_kernel<<<grid, 32, 0, stream>>>(R4, fc1p, (const float*)d_in[28],
                                            fc1out, nullptr, 1024, 16384, 1, 1);
  }
  // 4) FC2 -> fp32
  {
    dim3 grid(N_IMG / 16, 1);
    fc_wmma_kernel<<<grid, 32, 0, stream>>>(fc1out, fc2p, (const float*)d_in[30],
                                            nullptr, fc2buf, 6, 1024, 0, 0);
  }
  // 5) segment_sum -> d_out
  segment_kernel<<<1, 128, 0, stream>>>(fc2buf, index, (float*)d_out);
}
